// MultiScaleRetention_64733747085455
// MI455X (gfx1250) — matrix operand order repaired and checked by reading, not compiled, not run on hardware
//
#include <hip/hip_runtime.h>
#include <cmath>

#define B_   2
#define S_   2048
#define HID_ 512
#define H_   8
#define D_   64

typedef __bf16 bf16;
typedef __attribute__((ext_vector_type(16))) __bf16 v16bf;
typedef __attribute__((ext_vector_type(8)))  __bf16 v8bf;
typedef __attribute__((ext_vector_type(8)))  float   v8f;

union FragU { v16bf v; v8bf h[2]; };

// A-operand fragment (16x32 MxK, bf16): lane L holds row M=L%16;
// 8 contiguous elems at K0=(L/16)*8 and 8 more at K0+16 (ISA 7.12.2).
__device__ __forceinline__ v16bf frag_a(const bf16* base, int ld) {
  const int lane = threadIdx.x & 31;
  const bf16* p = base + (lane & 15) * ld + ((lane >> 4) << 3);
  FragU f;
  f.h[0] = *(const v8bf*)(p);
  f.h[1] = *(const v8bf*)(p + 16);
  return f.v;
}
// B-operand fragment from B^T storage ([N][K] row-major, bf16):
// lane L holds column N=L%16; 8 contiguous K at K0=(L/16)*8 and 8 more at K0+16.
__device__ __forceinline__ v16bf frag_b(const bf16* base, int ld) {
  const int lane = threadIdx.x & 31;
  const bf16* p = base + (lane & 15) * ld + ((lane >> 4) << 3);
  FragU f;
  f.h[0] = *(const v8bf*)(p);
  f.h[1] = *(const v8bf*)(p + 16);
  return f.v;
}
__device__ __forceinline__ v8f wmma_bf(v16bf a, v16bf b, v8f c) {
  return __builtin_amdgcn_wmma_f32_16x16x32_bf16(false, a, false, b, (short)0, c, false, false);
}
__device__ __forceinline__ v8f zero8() {
  v8f z;
#pragma unroll
  for (int i = 0; i < 8; ++i) z[i] = 0.f;
  return z;
}

// ---------------------------------------------------------------------------
// Kernel 1: pack complex weights -> bf16 B^T layouts.
//   wpack[h][j][e][d], j in {qr,qi,kr,ki,vr,vi}  (proj B^T: N=j*64+e, K=d)
//   w1t[n][k] = Re(w1[k][n]),  w2t[n][k] = Re(w2[k][n])
// ---------------------------------------------------------------------------
__global__ void k_pack(const float* wq, const float* wk, const float* wv,
                       const float* w1, const float* w2,
                       bf16* wpack, bf16* w1t, bf16* w2t) {
  const int NW = H_ * 6 * 64 * 64;  // 196608
  int idx = blockIdx.x * 256 + threadIdx.x;
  if (idx < NW) {
    int d = idx & 63;
    int e = (idx >> 6) & 63;
    int j = (idx >> 12) % 6;
    int h = idx / (6 * 4096);
    const float* src = (j < 2) ? wq : (j < 4) ? wk : wv;
    float val = src[(((h * 64 + d) * 64) + e) * 2 + (j & 1)];
    wpack[idx] = (bf16)val;
  } else if (idx < NW + 262144) {
    int t = idx - NW;
    int n = t >> 9, k = t & 511;
    w1t[t] = (bf16)w1[(k * 512 + n) * 2];
  } else if (idx < NW + 524288) {
    int t = idx - NW - 262144;
    int n = t >> 9, k = t & 511;
    w2t[t] = (bf16)w2[(k * 512 + n) * 2];
  }
}

// ---------------------------------------------------------------------------
// Kernel 2: per-head complex projections via WMMA.
//   A = Xh[s,d] (bf16), B^T = wpack rows (6*64 x 64).
//   qk[(bh*4+j)*S + s][e]   bf16 for j in {qr,qi,kr,ki}
//   vT[(bh*2+jv)*64 + e][s] bf16 for jv in {vr,vi} (pre-transposed B^T for S@V)
// grid(B*H, S/32), block 256 (8 waves: 2 row strips x 4 col quarters,
// 6 accumulator tiles per wave = 48 acc VGPRs -> no spills)
// ---------------------------------------------------------------------------
__global__ void k_proj(const float* __restrict__ x,
                       const bf16* __restrict__ wpack,
                       bf16* __restrict__ qk, bf16* __restrict__ vT) {
  __shared__ bf16 At[32][72];    //  4608 B
  __shared__ bf16 Bt[384][72];   // 55296 B
  const int tid = threadIdx.x;
  const int bh = blockIdx.x;
  const int b = bh / H_, h = bh % H_;
  const int s0 = blockIdx.y * 32;

  // stage weights (B^T) for this head
  for (int idx = tid; idx < 384 * 32; idx += 256) {
    int r = idx >> 5, c = idx & 31;
    ((unsigned*)&Bt[r][0])[c] =
        ((const unsigned*)(wpack + h * 6 * 4096))[r * 32 + c];
  }
  // stage Xh tile, f32 -> bf16
  for (int idx = tid; idx < 32 * 16; idx += 256) {
    int r = idx >> 4, c4 = idx & 15;
    float4 f = ((const float4*)(x + (size_t)((b * S_ + s0 + r) * HID_) + h * 64))[c4];
    bf16* dst = &At[r][c4 * 4];
    dst[0] = (bf16)f.x; dst[1] = (bf16)f.y; dst[2] = (bf16)f.z; dst[3] = (bf16)f.w;
  }
  __syncthreads();

  const int wid = tid >> 5;
  const int rowstrip = wid >> 2, colq = wid & 3;
  v8f acc[6];
#pragma unroll
  for (int t = 0; t < 6; ++t) acc[t] = zero8();

#pragma unroll
  for (int kk = 0; kk < 64; kk += 32) {
    v16bf fa = frag_a(&At[rowstrip * 16][kk], 72);
#pragma unroll
    for (int t = 0; t < 6; ++t) {
      v16bf fb = frag_b(&Bt[(colq * 6 + t) * 16][kk], 72);
      acc[t] = wmma_bf(fa, fb, acc[t]);
    }
  }

  const int lane = tid & 31;
#pragma unroll
  for (int t = 0; t < 6; ++t) {
    int cbase = (colq * 6 + t) * 16 + (lane & 15);
    int j = cbase >> 6, e = cbase & 63;
#pragma unroll
    for (int i = 0; i < 8; ++i) {
      int s = s0 + rowstrip * 16 + i + ((lane >> 4) << 3);
      float v = acc[t][i];
      if (j < 4)
        qk[(size_t)(((bh * 4 + j) * S_) + s) * 64 + e] = (bf16)v;
      else
        vT[(size_t)(((bh * 2 + (j - 4)) * 64) + e) * S_ + s] = (bf16)v;
    }
  }
}

// ---------------------------------------------------------------------------
// Kernel 3: decay-masked retention, streamed over causal m-blocks.
//   y[s, h*64+e] (f32) = [(qr kr^T - qi ki^T) o D] vr - [(qr ki^T + qi kr^T) o D] vi
// grid(B*H, S/64), block 128 (4 waves, one 16-row strip each)
// ---------------------------------------------------------------------------
__global__ void k_retention(const bf16* __restrict__ qk,
                            const bf16* __restrict__ vT,
                            float* __restrict__ y) {
  __shared__ bf16 krs[64][72], kis[64][72];       // 18432 B
  __shared__ bf16 vrs[64][72], vis[64][72];       // 18432 B
  __shared__ bf16 srs[4][16][72], sis[4][16][72]; // 18432 B

  const int tid = threadIdx.x;
  const int wid = tid >> 5, lane = tid & 31;
  const int bh = blockIdx.x;
  const int b = bh / H_, h = bh % H_;
  const int nb0 = blockIdx.y * 64;

  // gamma_h = 1 - exp(linspace(log(1/512), log(1/32), H))
  const float l0 = logf(1.f / 512.f), l1 = logf(1.f / 32.f);
  const float gamma = 1.f - expf(l0 + (l1 - l0) * ((float)h / (float)(H_ - 1)));
  const float lg2g = log2f(gamma);

  // q fragments for this wave's 16 rows, straight from global (L2-resident)
  const bf16* qrp = qk + (size_t)(((bh * 4 + 0) * S_) + nb0 + wid * 16) * 64;
  const bf16* qip = qk + (size_t)(((bh * 4 + 1) * S_) + nb0 + wid * 16) * 64;
  v16bf qr0 = frag_a(qrp, 64),      qr1 = frag_a(qrp + 32, 64);
  v16bf qi0 = frag_a(qip, 64),      qi1 = frag_a(qip + 32, 64);

  v8f accY[4];
#pragma unroll
  for (int t = 0; t < 4; ++t) accY[t] = zero8();

  const int nmb = blockIdx.y + 1;
  for (int mb = 0; mb < nmb; ++mb) {
    const int m0 = mb * 64;
    __syncthreads();  // previous iteration done with shared tiles
    // stage k (rows) and v^T (rows=e) tiles for this m-block
    const unsigned* krg = (const unsigned*)(qk + (size_t)(((bh * 4 + 2) * S_) + m0) * 64);
    const unsigned* kig = (const unsigned*)(qk + (size_t)(((bh * 4 + 3) * S_) + m0) * 64);
    for (int idx = tid; idx < 64 * 32; idx += 128) {
      int r = idx >> 5, c = idx & 31;
      ((unsigned*)&krs[r][0])[c] = krg[r * 32 + c];
      ((unsigned*)&kis[r][0])[c] = kig[r * 32 + c];
      ((unsigned*)&vrs[r][0])[c] =
          ((const unsigned*)(vT + (size_t)(((bh * 2 + 0) * 64) + r) * S_ + m0))[c];
      ((unsigned*)&vis[r][0])[c] =
          ((const unsigned*)(vT + (size_t)(((bh * 2 + 1) * 64) + r) * S_ + m0))[c];
    }
    __syncthreads();

    // score tiles: Sr = qr kr^T - qi ki^T ; Si = qr ki^T + qi kr^T, decayed
#pragma unroll
    for (int t = 0; t < 4; ++t) {
      v8f p = zero8(), q = zero8(), r = zero8();
#pragma unroll
      for (int kk = 0; kk < 64; kk += 32) {
        v16bf fkr = frag_b(&krs[t * 16][kk], 72);
        v16bf fki = frag_b(&kis[t * 16][kk], 72);
        v16bf ar = kk ? qr1 : qr0;
        v16bf ai = kk ? qi1 : qi0;
        p = wmma_bf(ar, fkr, p);
        q = wmma_bf(ai, fki, q);
        r = wmma_bf(ar, fki, r);
        r = wmma_bf(ai, fkr, r);
      }
#pragma unroll
      for (int i = 0; i < 8; ++i) {
        int n = nb0 + wid * 16 + i + ((lane >> 4) << 3);
        int m = m0 + t * 16 + (lane & 15);
        float d = (n >= m) ? exp2f((float)(n - m) * lg2g) : 0.f;
        int rr = i + ((lane >> 4) << 3), cc = t * 16 + (lane & 15);
        srs[wid][rr][cc] = (bf16)((p[i] - q[i]) * d);
        sis[wid][rr][cc] = (bf16)(-(r[i]) * d);  // pre-negated Si
      }
    }
    // per-wave LDS ops are in order; no barrier needed before reusing own tile
#pragma unroll
    for (int kk = 0; kk < 64; kk += 32) {
      v16bf fsr = frag_a(&srs[wid][0][kk], 72);
      v16bf fsi = frag_a(&sis[wid][0][kk], 72);
#pragma unroll
      for (int et = 0; et < 4; ++et) {
        v16bf fvr = frag_b(&vrs[et * 16][kk], 72);
        v16bf fvi = frag_b(&vis[et * 16][kk], 72);
        accY[et] = wmma_bf(fsr, fvr, accY[et]);
        accY[et] = wmma_bf(fsi, fvi, accY[et]);
      }
    }
  }

#pragma unroll
  for (int et = 0; et < 4; ++et)
#pragma unroll
    for (int i = 0; i < 8; ++i) {
      int s = nb0 + wid * 16 + i + ((lane >> 4) << 3);
      int e = et * 16 + (lane & 15);
      y[(size_t)((b * S_ + s) * HID_) + h * 64 + e] = accY[et][i];
    }
}

// ---------------------------------------------------------------------------
// Kernel 4: GroupNorm (groups == heads, 64 ch each) per token. One wave/group.
// ---------------------------------------------------------------------------
__global__ void k_gnorm(const float* __restrict__ y,
                        const float* __restrict__ gns, const float* __restrict__ gnb,
                        float* __restrict__ yn) {
  const int lane = threadIdx.x & 31;
  const int gid = blockIdx.x * 8 + (threadIdx.x >> 5);  // 0 .. B*S*H-1
  const int h = gid % H_;
  const int tok = gid / H_;
  const size_t base = (size_t)tok * HID_ + h * 64;
  float v0 = y[base + lane], v1 = y[base + lane + 32];
  float s = v0 + v1, ss = v0 * v0 + v1 * v1;
#pragma unroll
  for (int off = 16; off > 0; off >>= 1) {
    s  += __shfl_xor(s, off);
    ss += __shfl_xor(ss, off);
  }
  float mu = s * (1.f / 64.f);
  float var = ss * (1.f / 64.f) - mu * mu;
  float inv = rsqrtf(var + 1e-5f);
  int c = h * 64 + lane;
  yn[base + lane]      = (v0 - mu) * inv * gns[c]      + gnb[c];
  yn[base + lane + 32] = (v1 - mu) * inv * gns[c + 32] + gnb[c + 32];
}

// ---------------------------------------------------------------------------
// Kernel 5/6: 4096x512x512 bf16 WMMA GEMM.
//   mode 0: A = x (f32 -> bf16), epilogue swish(g)*yn -> bf16 t
//   mode 1: A = t (bf16),        epilogue -> f32 out
// grid(32, 8), block 256 (8 waves x 16 rows; 64 N-cols per block)
// ---------------------------------------------------------------------------
__global__ void k_gemm(const float* __restrict__ xsrc,
                       const bf16* __restrict__ absrc,
                       const bf16* __restrict__ wt,
                       const float* __restrict__ yn,
                       bf16* __restrict__ tout, float* __restrict__ fout,
                       int mode) {
  __shared__ bf16 Asm[128][72];  // 18432 B
  __shared__ bf16 Bsm[64][72];   //  9216 B
  const int tid = threadIdx.x;
  const int bm = blockIdx.x, bn = blockIdx.y;
  const int wid = tid >> 5, lane = tid & 31;

  v8f acc[4];
#pragma unroll
  for (int t = 0; t < 4; ++t) acc[t] = zero8();

  for (int kb = 0; kb < 8; ++kb) {
    __syncthreads();
    if (mode == 0) {
      for (int idx = tid; idx < 128 * 16; idx += 256) {
        int r = idx >> 4, c4 = idx & 15;
        float4 f = ((const float4*)(xsrc + (size_t)(bm * 128 + r) * 512 + kb * 64))[c4];
        bf16* dst = &Asm[r][c4 * 4];
        dst[0] = (bf16)f.x; dst[1] = (bf16)f.y; dst[2] = (bf16)f.z; dst[3] = (bf16)f.w;
      }
    } else {
      for (int idx = tid; idx < 128 * 8; idx += 256) {
        int r = idx >> 3, c = idx & 7;
        *(int4*)&Asm[r][c * 8] =
            ((const int4*)(absrc + (size_t)(bm * 128 + r) * 512 + kb * 64))[c];
      }
    }
    for (int idx = tid; idx < 64 * 8; idx += 256) {
      int r = idx >> 3, c = idx & 7;
      *(int4*)&Bsm[r][c * 8] =
          ((const int4*)(wt + (size_t)(bn * 64 + r) * 512 + kb * 64))[c];
    }
    if (kb + 1 < 8) {  // warm L2/L0 for next K chunk
      if (mode == 0)
        __builtin_prefetch(xsrc + (size_t)(bm * 128 + (tid >> 1)) * 512 + (kb + 1) * 64, 0, 1);
      else
        __builtin_prefetch(absrc + (size_t)(bm * 128 + tid / 2) * 512 + (kb + 1) * 64, 0, 1);
    }
    __syncthreads();

#pragma unroll
    for (int kk = 0; kk < 64; kk += 32) {
      v16bf fa = frag_a(&Asm[wid * 16][kk], 72);
#pragma unroll
      for (int t = 0; t < 4; ++t) {
        v16bf fb = frag_b(&Bsm[t * 16][kk], 72);
        acc[t] = wmma_bf(fa, fb, acc[t]);
      }
    }
  }

#pragma unroll
  for (int t = 0; t < 4; ++t)
#pragma unroll
    for (int i = 0; i < 8; ++i) {
      int row = bm * 128 + wid * 16 + i + ((lane >> 4) << 3);
      int col = bn * 64 + t * 16 + (lane & 15);
      float g = acc[t][i];
      if (mode == 0) {
        float sw = g / (1.f + expf(-g));              // swish
        tout[(size_t)row * 512 + col] = (bf16)(sw * yn[(size_t)row * 512 + col]);
      } else {
        fout[(size_t)row * 512 + col] = g;
      }
    }
}

// ---------------------------------------------------------------------------
extern "C" void kernel_launch(void* const* d_in, const int* in_sizes, int n_in,
                              void* d_out, int out_size, void* d_ws, size_t ws_size,
                              hipStream_t stream) {
  const float* x   = (const float*)d_in[0];
  const float* wq  = (const float*)d_in[1];  // complex64 interleaved
  const float* wk  = (const float*)d_in[2];
  const float* wv  = (const float*)d_in[3];
  const float* w1  = (const float*)d_in[4];
  const float* w2  = (const float*)d_in[5];
  const float* gns = (const float*)d_in[6];
  const float* gnb = (const float*)d_in[7];
  float* out = (float*)d_out;

  char* ws = (char*)d_ws;
  size_t off = 0;
  auto alloc = [&](size_t bytes) -> void* {
    void* p = ws + off;
    off += (bytes + 255) & ~(size_t)255;
    return p;
  };
  bf16*  wpack = (bf16*)alloc((size_t)H_ * 6 * 64 * 64 * sizeof(bf16));
  bf16*  w1t   = (bf16*)alloc((size_t)512 * 512 * sizeof(bf16));
  bf16*  w2t   = (bf16*)alloc((size_t)512 * 512 * sizeof(bf16));
  bf16*  qk    = (bf16*)alloc((size_t)B_ * H_ * 4 * S_ * 64 * sizeof(bf16));
  bf16*  vT    = (bf16*)alloc((size_t)B_ * H_ * 2 * 64 * S_ * sizeof(bf16));
  float* ybuf  = (float*)alloc((size_t)B_ * S_ * HID_ * sizeof(float));
  float* ynbuf = (float*)alloc((size_t)B_ * S_ * HID_ * sizeof(float));
  bf16*  tbuf  = (bf16*)alloc((size_t)B_ * S_ * HID_ * sizeof(bf16));
  (void)ws_size; (void)in_sizes; (void)n_in; (void)out_size;

  const int packTot = H_ * 6 * 64 * 64 + 2 * 512 * 512;
  k_pack<<<(packTot + 255) / 256, 256, 0, stream>>>(wq, wk, wv, w1, w2, wpack, w1t, w2t);
  k_proj<<<dim3(B_ * H_, S_ / 32), 256, 0, stream>>>(x, wpack, qk, vT);
  k_retention<<<dim3(B_ * H_, S_ / 64), 128, 0, stream>>>(qk, vT, ybuf);
  k_gnorm<<<(B_ * S_ * H_) / 8, 256, 0, stream>>>(ybuf, gns, gnb, ynbuf);
  k_gemm<<<dim3(32, 8), 256, 0, stream>>>(x, nullptr, w1t, ynbuf, tbuf, nullptr, 0);
  k_gemm<<<dim3(32, 8), 256, 0, stream>>>(nullptr, tbuf, w2t, nullptr, nullptr, out, 1);
}
